// NSF_prior_80633716015312
// MI455X (gfx1250) — matrix operand order repaired and checked by reading, not compiled, not run on hardware
//
#include <hip/hip_runtime.h>
#include <math.h>

typedef __attribute__((ext_vector_type(16))) _Float16 v16h;
typedef __attribute__((ext_vector_type(8)))  _Float16 v8h;
typedef __attribute__((ext_vector_type(8)))  float    v8f;

#define NPTS 2097152
#define NB 20
#define LOG2PI 1.8378770664093453f

// ---------- small helpers ----------
__device__ __forceinline__ v16h cat16(v8h lo, v8h hi) {
  return __builtin_shufflevector(lo, hi, 0,1,2,3,4,5,6,7,8,9,10,11,12,13,14,15);
}

__device__ __forceinline__ float softplusf(float x) {
  return (x > 20.f) ? x : log1pf(__expf(x));
}

__device__ __forceinline__ v8f wmma16(v16h a, v16h b, v8f c) {
  // D = A(16x32 f16) * B(32x16 f16) + C(16x16 f32)
  return __builtin_amdgcn_wmma_f32_16x16x32_f16(false, a, false, b, (short)0, c, false, false);
}

// A-fragment gather: 16-bit A 16x32 layout (lanes 0-15: K 0..7 & 16..23;
// lanes 16-31: K 8..15 & 24..31), row M = lane&15. Two 16B LDS reads.
__device__ __forceinline__ v16h loadA(const _Float16* tile, int stride, int mt, int kc, int lane) {
  int row = mt * 16 + (lane & 15);
  int c0  = kc * 32 + ((lane & 16) ? 8 : 0);
  const _Float16* p = tile + row * stride + c0;
  return cat16(*(const v8h*)p, *(const v8h*)(p + 16));
}

// B-fragment gather from n-major transposed weights wT[n][kPad]:
// lanes 0-15 hold K 0..7 & 16..23, lanes 16-31 hold K 8..15 & 24..31, N = lane&15.
__device__ __forceinline__ v16h loadB(const _Float16* wT, int kPad, int nt, int kc, int lane) {
  int n  = (lane & 15) + nt * 16;
  int kb = kc * 32 + ((lane & 16) ? 8 : 0);
  const _Float16* p = wT + n * kPad + kb;
  return cat16(*(const v8h*)p, *(const v8h*)(p + 16));
}

// C/D layout: VGPR i -> row M = i + (lane<16 ? 0 : 8), col N = lane&15.
__device__ __forceinline__ void store_act(v8f acc, _Float16* h, int stride, int mt, int nt,
                                          const float* bias, int lane) {
  int n  = nt * 16 + (lane & 15);
  float b = bias[n];
  int r0 = mt * 16 + ((lane & 16) ? 8 : 0);
#pragma unroll
  for (int i = 0; i < 8; ++i) {
    float v = acc[i] + b;
    v = (v > 0.f) ? v : 0.01f * v;           // leaky_relu(0.01)
    h[(r0 + i) * stride + n] = (_Float16)v;
  }
}

__device__ __forceinline__ void store_tp(v8f acc, float* tp, int stride, int mt, int nt,
                                         const float* bias, int realN, int lane) {
  int n = nt * 16 + (lane & 15);
  if (n < realN) {
    float b = bias[n];
    int r0 = mt * 16 + ((lane & 16) ? 8 : 0);
#pragma unroll
    for (int i = 0; i < 8; ++i) tp[(r0 + i) * stride + n] = acc[i] + b;
  }
}

// stage weight matrix src[kReal][nReal] (row-major) -> LDS transposed [nDst][kPad], zero padded
__device__ void stageT(_Float16* dst, const float* src, int nDst, int kPad, int kReal, int nReal) {
  for (int idx = threadIdx.x; idx < nDst * kPad; idx += 256) {
    int n = idx / kPad, k = idx - n * kPad;
    float v = (k < kReal && n < nReal) ? src[k * nReal + n] : 0.f;
    dst[idx] = (_Float16)v;
  }
}

// ---------- per-wave 3-layer MLP (32 rows): xin[32x64] -> tp[32x59] ----------
__device__ __forceinline__ void mlp3_wave(const _Float16* xin, _Float16* h, float* tp,
                                          const _Float16* w0T, const _Float16* w1T, const _Float16* w2T,
                                          const float* b0, const float* b1, const float* b2, int lane) {
  // L1: K=64 (39 padded), N=32, leaky
#pragma unroll
  for (int mt = 0; mt < 2; ++mt) {
    v16h a0 = loadA(xin, 72, mt, 0, lane);
    v16h a1 = loadA(xin, 72, mt, 1, lane);
#pragma unroll
    for (int nt = 0; nt < 2; ++nt) {
      v8f acc = {0.f,0.f,0.f,0.f,0.f,0.f,0.f,0.f};
      acc = wmma16(a0, loadB(w0T, 64, nt, 0, lane), acc);
      acc = wmma16(a1, loadB(w0T, 64, nt, 1, lane), acc);
      store_act(acc, h, 40, mt, nt, b0, lane);
    }
  }
  __builtin_amdgcn_wave_barrier();
  // L2: K=32, N=32, leaky (in place; A consumed into regs before stores)
#pragma unroll
  for (int mt = 0; mt < 2; ++mt) {
    v16h a = loadA(h, 40, mt, 0, lane);
#pragma unroll
    for (int nt = 0; nt < 2; ++nt) {
      v8f acc = {0.f,0.f,0.f,0.f,0.f,0.f,0.f,0.f};
      acc = wmma16(a, loadB(w1T, 32, nt, 0, lane), acc);
      store_act(acc, h, 40, mt, nt, b1, lane);
    }
  }
  __builtin_amdgcn_wave_barrier();
  // L3: K=32, N=64 (59 real), fp32 out + bias
#pragma unroll
  for (int mt = 0; mt < 2; ++mt) {
    v16h a = loadA(h, 40, mt, 0, lane);
#pragma unroll
    for (int nt = 0; nt < 4; ++nt) {
      v8f acc = {0.f,0.f,0.f,0.f,0.f,0.f,0.f,0.f};
      acc = wmma16(a, loadB(w2T, 32, nt, 0, lane), acc);
      store_tp(acc, tp, 61, mt, nt, b2, 59, lane);
    }
  }
  __builtin_amdgcn_wave_barrier();
}

// ---------- per-lane rational quadratic spline ----------
__device__ __forceinline__ void rqs(float x, const float* tp, float* yOut, float* ladOut) {
  float mw = -1e30f, mh = -1e30f;
  for (int i = 0; i < NB; ++i) { mw = fmaxf(mw, tp[i]); mh = fmaxf(mh, tp[NB + i]); }
  float sw = 0.f, sh = 0.f;
  for (int i = 0; i < NB; ++i) { sw += __expf(tp[i] - mw); sh += __expf(tp[NB + i] - mh); }
  const float aw = (1.f - 0.001f * NB) / sw;
  const float ah = (1.f - 0.001f * NB) / sh;
  float xc = fminf(fmaxf(x, -1.f), 1.f);
  float cl = -1.f, hl = -1.f;
  float in_cw = -1.f, in_ch = -1.f, in_w = 1.f, in_h = 1.f;
  int idx = 0;
  for (int i = 0; i < NB; ++i) {
    float wi = 2.f * (0.001f + aw * __expf(tp[i] - mw));
    float hi = 2.f * (0.001f + ah * __expf(tp[NB + i] - mh));
    float cr = (i == NB - 1) ? 1.f : cl + wi;
    float hr = (i == NB - 1) ? 1.f : hl + hi;
    if (xc >= cl) { idx = i; in_cw = cl; in_ch = hl; in_w = cr - cl; in_h = hr - hl; }
    cl = cr; hl = hr;
  }
  // d = MIN_D + softplus(ud); boundary knots are exactly 1.0 by construction
  float da = (idx == 0)      ? 1.f : 0.001f + softplusf(tp[2 * NB + idx - 1]);
  float db = (idx == NB - 1) ? 1.f : 0.001f + softplusf(tp[2 * NB + idx]);
  float th  = (xc - in_cw) / in_w;
  float omt = 1.f - th;
  float t1  = th * omt;
  float dl  = in_h / in_w;
  float num = in_h * (dl * th * th + da * t1);
  float den = dl + (da + db - 2.f * dl) * t1;
  float y   = in_ch + num / den;
  float dnum = dl * dl * (db * th * th + 2.f * dl * t1 + da * omt * omt);
  float lad  = __logf(dnum) - 2.f * __logf(den);
  bool inside = (x >= -1.f) && (x <= 1.f);
  *yOut   = inside ? y : x;
  *ladOut = inside ? lad : 0.f;
}

__device__ __forceinline__ void pe9(_Float16* xrow, float y) {
  // pe(y, 4 freqs, linear): freqs = linspace(1, 8, 4)
  const float fr0 = 1.f, fr1 = 10.f / 3.f, fr2 = 17.f / 3.f, fr3 = 8.f;
  xrow[0] = (_Float16)y;
  xrow[1] = (_Float16)__sinf(y * fr0); xrow[2] = (_Float16)__cosf(y * fr0);
  xrow[3] = (_Float16)__sinf(y * fr1); xrow[4] = (_Float16)__cosf(y * fr1);
  xrow[5] = (_Float16)__sinf(y * fr2); xrow[6] = (_Float16)__cosf(y * fr2);
  xrow[7] = (_Float16)__sinf(y * fr3); xrow[8] = (_Float16)__cosf(y * fr3);
}

// ---------- kernel ----------
__global__ void __launch_bounds__(256)
nsf_prior_kernel(const float* __restrict__ wi,   const float* __restrict__ cond,
                 const float* __restrict__ p0_w0, const float* __restrict__ p0_b0,
                 const float* __restrict__ p0_w1, const float* __restrict__ p0_b1,
                 const float* __restrict__ m1_w0, const float* __restrict__ m1_b0,
                 const float* __restrict__ m1_w1, const float* __restrict__ m1_b1,
                 const float* __restrict__ m1_w2, const float* __restrict__ m1_b2,
                 const float* __restrict__ m0_w0, const float* __restrict__ m0_b0,
                 const float* __restrict__ m0_w1, const float* __restrict__ m0_b1,
                 const float* __restrict__ m0_w2, const float* __restrict__ m0_b2,
                 float* __restrict__ out) {
  extern __shared__ __attribute__((aligned(16))) char smem_raw[];
  _Float16* W = (_Float16*)smem_raw;
  _Float16* w0T_1 = W + 0;      // 32 x 64
  _Float16* w1T_1 = W + 2048;   // 32 x 32
  _Float16* w2T_1 = W + 3072;   // 64 x 32
  _Float16* w0T_0 = W + 5120;
  _Float16* w1T_0 = W + 7168;
  _Float16* w2T_0 = W + 8192;
  _Float16* pw0T  = W + 10240;  // 32 x 32
  _Float16* pw1T  = W + 11264;  // 16 x 32

  stageT(w0T_1, m1_w0, 32, 64, 39, 32);
  stageT(w1T_1, m1_w1, 32, 32, 32, 32);
  stageT(w2T_1, m1_w2, 64, 32, 32, 59);
  stageT(w0T_0, m0_w0, 32, 64, 39, 32);
  stageT(w1T_0, m0_w1, 32, 32, 32, 32);
  stageT(w2T_0, m0_w2, 64, 32, 32, 59);
  stageT(pw0T,  p0_w0, 32, 32, 30, 32);
  stageT(pw1T,  p0_w1, 16, 32, 32, 4);
  __syncthreads();

  const int tid = threadIdx.x;
  const int wave = tid >> 5, lane = tid & 31;
  char* wb = smem_raw + 23552 + wave * 17536;
  _Float16* xinT = (_Float16*)wb;      // 32 x 72 f16
  _Float16* hT   = xinT + 2304;        // 32 x 40 f16
  _Float16* cT   = hT + 1280;          // 32 x 40 f16 (cond_e)
  float*    tpT  = (float*)(cT + 1280);// 32 x 61 f32

  const int r = blockIdx.x * 256 + wave * 32 + lane;
  const float x0 = wi[2 * r], x1 = wi[2 * r + 1];

  // cond_e = [pe(cond[8:10], 5 freqs log) , cond[0:8]]  (30 values)
  {
    _Float16* crow = cT + lane * 40;
    _Float16* xrow = xinT + lane * 72;
    float c8 = cond[r * 10 + 8], c9 = cond[r * 10 + 9];
    float e[30];
    e[0] = c8; e[1] = c9;
    float f = 1.f; int o = 2;
#pragma unroll
    for (int i = 0; i < 5; ++i) {
      e[o++] = __sinf(c8 * f); e[o++] = __sinf(c9 * f);
      e[o++] = __cosf(c8 * f); e[o++] = __cosf(c9 * f);
      f *= 2.f;
    }
#pragma unroll
    for (int i = 0; i < 8; ++i) e[22 + i] = cond[r * 10 + i];
#pragma unroll
    for (int j = 0; j < 30; ++j) {
      _Float16 hv = (_Float16)e[j];
      crow[j] = hv; xrow[9 + j] = hv;
    }
    crow[30] = (_Float16)0.f; crow[31] = (_Float16)0.f;
    for (int j = 39; j < 64; ++j) xrow[j] = (_Float16)0.f;
    pe9(xrow, x1);
  }
  __builtin_amdgcn_wave_barrier();

  // flow 1: transform wi0 conditioned on pe(wi1)+cond_e
  mlp3_wave(xinT, hT, tpT, w0T_1, w1T_1, w2T_1, m1_b0, m1_b1, m1_b2, lane);
  float y0, ld0;
  rqs(x0, tpT + lane * 61, &y0, &ld0);

  // flow 2: transform wi1 conditioned on pe(y0)+cond_e
  pe9(xinT + lane * 72, y0);
  __builtin_amdgcn_wave_barrier();
  mlp3_wave(xinT, hT, tpT, w0T_0, w1T_0, w2T_0, m0_b0, m0_b1, m0_b2, lane);
  float y1, ld1;
  rqs(x1, tpT + lane * 61, &y1, &ld1);
  __builtin_amdgcn_wave_barrier();

  // prior MLP p0: cond_e(30 pad 32) -> 32 leaky -> 4  (mu, log_sigma)
#pragma unroll
  for (int mt = 0; mt < 2; ++mt) {
    v16h a = loadA(cT, 40, mt, 0, lane);
#pragma unroll
    for (int nt = 0; nt < 2; ++nt) {
      v8f acc = {0.f,0.f,0.f,0.f,0.f,0.f,0.f,0.f};
      acc = wmma16(a, loadB(pw0T, 32, nt, 0, lane), acc);
      store_act(acc, hT, 40, mt, nt, p0_b0, lane);
    }
  }
  __builtin_amdgcn_wave_barrier();
#pragma unroll
  for (int mt = 0; mt < 2; ++mt) {
    v16h a = loadA(hT, 40, mt, 0, lane);
    v8f acc = {0.f,0.f,0.f,0.f,0.f,0.f,0.f,0.f};
    acc = wmma16(a, loadB(pw1T, 32, 0, 0, lane), acc);
    store_tp(acc, tpT, 61, mt, 0, p0_b1, 4, lane);
  }
  __builtin_amdgcn_wave_barrier();

  const float* my = tpT + lane * 61;
  float mu0 = my[0], mu1 = my[1], ls0 = my[2], ls1 = my[3];
  float z0 = (y0 - mu0) * __expf(-ls0);
  float z1 = (y1 - mu1) * __expf(-ls1);
  float lp = -(ls0 + ls1) - LOG2PI - 0.5f * (z0 * z0 + z1 * z1) + ld0 + ld1;

  out[2 * r]     = y0;
  out[2 * r + 1] = y1;
  out[2 * NPTS + r] = lp;
}

extern "C" void kernel_launch(void* const* d_in, const int* in_sizes, int n_in,
                              void* d_out, int out_size, void* d_ws, size_t ws_size,
                              hipStream_t stream) {
  const float* wi    = (const float*)d_in[0];
  const float* cond  = (const float*)d_in[1];
  const float* p0_w0 = (const float*)d_in[2];
  const float* p0_b0 = (const float*)d_in[3];
  const float* p0_w1 = (const float*)d_in[4];
  const float* p0_b1 = (const float*)d_in[5];
  const float* m1_w0 = (const float*)d_in[6];
  const float* m1_b0 = (const float*)d_in[7];
  const float* m1_w1 = (const float*)d_in[8];
  const float* m1_b1 = (const float*)d_in[9];
  const float* m1_w2 = (const float*)d_in[10];
  const float* m1_b2 = (const float*)d_in[11];
  const float* m0_w0 = (const float*)d_in[12];
  const float* m0_b0 = (const float*)d_in[13];
  const float* m0_w1 = (const float*)d_in[14];
  const float* m0_b1 = (const float*)d_in[15];
  const float* m0_w2 = (const float*)d_in[16];
  const float* m0_b2 = (const float*)d_in[17];

  const size_t shmem = 23552 + 8 * 17536; // 160 KB dynamic LDS (WGP has 320 KB)
  (void)hipFuncSetAttribute((const void*)nsf_prior_kernel,
                            hipFuncAttributeMaxDynamicSharedMemorySize, (int)shmem);
  dim3 grid(NPTS / 256), block(256);
  hipLaunchKernelGGL(nsf_prior_kernel, grid, block, shmem, stream,
                     wi, cond, p0_w0, p0_b0, p0_w1, p0_b1,
                     m1_w0, m1_b0, m1_w1, m1_b1, m1_w2, m1_b2,
                     m0_w0, m0_b0, m0_w1, m0_b1, m0_w2, m0_b2,
                     (float*)d_out);
}
